// EEGFormer_20529943675128
// MI455X (gfx1250) — matrix operand order repaired and checked by reading, not compiled, not run on hardware
//
#include <hip/hip_runtime.h>
#include <hip/hip_bf16.h>

// ---------------------------------------------------------------------------
// EEGFormer block for MI455X (gfx1250, wave32, WMMA).
// Compute-bound (~223 GFLOP vs ~0.4 GB traffic) -> everything through
// v_wmma_f32_16x16x32_bf16 with f32 accumulation.
// Round 2: flash-style attention (halved score-register footprint, no
// s_set_vgpr_msb churn) and 2x WMMA density per B-fragment in the GEMM.
// ---------------------------------------------------------------------------

typedef __attribute__((ext_vector_type(16))) __bf16 v16bf;
typedef __attribute__((ext_vector_type(8)))  float  v8f;

union V16U { uint4 u[2]; v16bf v; };
union V16S { __bf16 e[16]; v16bf v; };

__device__ __forceinline__ v8f wmma_bf16(v16bf a, v16bf b, v8f c) {
  // 8 args: (neg_a, A, neg_b, B, c_mod, C, reuse_a, reuse_b)
  return __builtin_amdgcn_wmma_f32_16x16x32_bf16(false, a, false, b, (short)0, c,
                                                 false, false);
}

// A-fragment (16x32, MxK): lane row = lane&15; elements 0..15 -> K = kb..kb+15,
// kb = (lane>>4)*16. Two 16B loads.
__device__ __forceinline__ v16bf load_afrag(const __bf16* base, int ld, int lane) {
  const int row = lane & 15;
  const int kb  = (lane >> 4) << 4;
  const __bf16* p = base + row * ld + kb;
  V16U r;
  r.u[0] = *reinterpret_cast<const uint4*>(p);
  r.u[1] = *reinterpret_cast<const uint4*>(p + 8);
  return r.v;
}

// B-fragment (32x16, KxN): lane col = lane&15; elements 0..15 are 16
// contiguous K values (K offset 16*(lane>>4) handled by caller).
__device__ __forceinline__ v16bf load_bfrag16(const __bf16* p) {
  V16U r;
  r.u[0] = *reinterpret_cast<const uint4*>(p);
  r.u[1] = *reinterpret_cast<const uint4*>(p + 8);
  return r.v;
}

// ---------------------------------------------------------------------------
// f32 -> bf16 conversion (weights)
// ---------------------------------------------------------------------------
__global__ __launch_bounds__(256) void cvt_f32_bf16(const float* __restrict__ in,
                                                    __bf16* __restrict__ out, int n) {
  int i = (blockIdx.x * 256 + threadIdx.x) * 4;
  if (i + 3 < n) {
    float4 f = *reinterpret_cast<const float4*>(in + i);
    out[i + 0] = (__bf16)f.x;
    out[i + 1] = (__bf16)f.y;
    out[i + 2] = (__bf16)f.z;
    out[i + 3] = (__bf16)f.w;
  }
}

// ---------------------------------------------------------------------------
// LayerNorm over L=512, one wave per row, shfl_xor reductions, bf16 out.
// ---------------------------------------------------------------------------
__global__ __launch_bounds__(128) void ln_bf16(const float* __restrict__ x,
                                               const float* __restrict__ g,
                                               const float* __restrict__ be,
                                               __bf16* __restrict__ out) {
  const int lane = threadIdx.x & 31;
  const int wave = threadIdx.x >> 5;
  const size_t row = (size_t)blockIdx.x * 4 + wave;
  const float* xr = x + row * 512;
  float v[16];
  float s = 0.f, ss = 0.f;
#pragma unroll
  for (int i = 0; i < 16; ++i) {
    v[i] = xr[lane + i * 32];
    s += v[i];
    ss += v[i] * v[i];
  }
#pragma unroll
  for (int m = 1; m < 32; m <<= 1) {
    s  += __shfl_xor(s, m);
    ss += __shfl_xor(ss, m);
  }
  const float mu  = s * (1.0f / 512.0f);
  const float var = ss * (1.0f / 512.0f) - mu * mu;
  const float r   = rsqrtf(var + 1e-5f);
  __bf16* orow = out + row * 512;
#pragma unroll
  for (int i = 0; i < 16; ++i) {
    const int c = lane + i * 32;
    orow[c] = (__bf16)((v[i] - mu) * r * g[c] + be[c]);
  }
}

// ---------------------------------------------------------------------------
// Generic GEMM: out = act(A[M,K] * W[N,K]^T + bias) [+ res].
// Block = 256 threads (8 waves), tile 256x64, K-chunks of 32.
// Each wave owns 32 rows (2 A-frags) x 64 cols -> 2 WMMA per B-fragment.
// A staged in LDS; B fragments straight from global (L2-resident weights).
// ---------------------------------------------------------------------------
template <typename OutT, bool RELU>
__global__ __launch_bounds__(256) void gemm_wmma(
    const __bf16* __restrict__ A, const __bf16* __restrict__ W,
    const float* __restrict__ bias, const float* __restrict__ res,
    OutT* __restrict__ out, int M, int N, int K) {
  __shared__ __align__(16) __bf16 sA[256 * 32];
  const int tid  = threadIdx.x;
  const int lane = tid & 31;
  const int wave = tid >> 5;
  const int half = lane >> 4;
  const int l15  = lane & 15;
  const int nb   = N >> 6;
  const int m0   = (blockIdx.x / nb) * 256;
  const int n0   = (blockIdx.x % nb) * 64;

  v8f acc[4][2] = {};

  // cooperative A staging: one 256-row tile, 32 bf16 (64B) per thread-row
  const __bf16* aRow = A + (size_t)(m0 + tid) * K;

  for (int k0 = 0; k0 < K; k0 += 32) {
    __syncthreads();
    const uint4* src = reinterpret_cast<const uint4*>(aRow + k0);
    uint4* dst = reinterpret_cast<uint4*>(sA + tid * 32);
    dst[0] = src[0];
    dst[1] = src[1];
    dst[2] = src[2];
    dst[3] = src[3];
    if (k0 + 32 < K) __builtin_prefetch(aRow + k0 + 32, 0, 1);
    __syncthreads();
    v16bf a0 = load_afrag(sA + (wave * 32 + 0) * 32, 32, lane);
    v16bf a1 = load_afrag(sA + (wave * 32 + 16) * 32, 32, lane);
#pragma unroll
    for (int n = 0; n < 4; ++n) {
      const __bf16* bp = W + (size_t)(n0 + n * 16 + l15) * K + k0 + half * 16;
      v16bf b = load_bfrag16(bp);
      acc[n][0] = wmma_bf16(a0, b, acc[n][0]);
      acc[n][1] = wmma_bf16(a1, b, acc[n][1]);
    }
  }

#pragma unroll
  for (int n = 0; n < 4; ++n) {
    const int col  = n0 + n * 16 + l15;
    const float bz = bias ? bias[col] : 0.0f;
#pragma unroll
    for (int u = 0; u < 2; ++u)
#pragma unroll
      for (int j = 0; j < 8; ++j) {
        const int row = m0 + wave * 32 + u * 16 + half * 8 + j;
        float v = acc[n][u][j] + bz;
        if (RELU) v = fmaxf(v, 0.0f);
        const size_t idx = (size_t)row * N + col;
        if (res) v += res[idx];
        out[idx] = (OutT)v;
      }
  }
}

// ---------------------------------------------------------------------------
// Regional attention (flash-style): one block per (b,h,s). C=256, D=64.
// 4 waves; each wave owns a 16-query stripe per iteration (4 iterations).
// Keys processed in 2 blocks of 128 with online softmax (running max/sum,
// rescaled O accumulators) -> only 8 score tiles (64 VGPRs) live at a time.
// P transposed through LDS into A-fragment layout for P*V.
// ---------------------------------------------------------------------------
__global__ __launch_bounds__(128) void attn_wmma(const __bf16* __restrict__ Q,
                                                 const __bf16* __restrict__ Kt,
                                                 const __bf16* __restrict__ V,
                                                 __bf16* __restrict__ O) {
  __shared__ __align__(16) __bf16 sP[4][16 * 128];
  const int tid  = threadIdx.x;
  const int lane = tid & 31;
  const int wave = tid >> 5;
  const int half = lane >> 4;
  const int l15  = lane & 15;
  const int blk  = blockIdx.x;  // b*256 + h*32 + s
  const int s    = blk & 31;
  const int h    = (blk >> 5) & 7;
  const int b    = blk >> 8;
  const size_t tokBase = (size_t)(b * 32 + s) * 256;
  const int fb = h * 64;

  for (int qs = 0; qs < 4; ++qs) {
    const int q0 = (qs * 4 + wave) * 16;

    // Q fragments for both 32-wide K chunks of D=64
    v16bf aQ[2];
#pragma unroll
    for (int kc = 0; kc < 2; ++kc)
      aQ[kc] = load_afrag(Q + (tokBase + q0) * 512 + fb + kc * 32, 512, lane);

    float mrow[8], lrow[8];
#pragma unroll
    for (int j = 0; j < 8; ++j) {
      mrow[j] = -3.0e38f;
      lrow[j] = 0.0f;
    }
    v8f oacc[4] = {};
    __bf16* pRow = &sP[wave][0];

    for (int kb = 0; kb < 2; ++kb) {
      const size_t keyBase = tokBase + kb * 128;

      // scores for this 128-key block -> 8 accumulator tiles
      v8f sc[8] = {};
#pragma unroll
      for (int t = 0; t < 8; ++t) {
#pragma unroll
        for (int kc = 0; kc < 2; ++kc) {
          const __bf16* bp =
              Kt + (keyBase + t * 16 + l15) * 512 + fb + kc * 32 + half * 16;
          sc[t] = wmma_bf16(aQ[kc], load_bfrag16(bp), sc[t]);
        }
      }

      // online softmax: row r = half*8 + j over the 16 lanes of this half
      float fj[8];
#pragma unroll
      for (int j = 0; j < 8; ++j) {
        float bm = -3.0e38f;
#pragma unroll
        for (int t = 0; t < 8; ++t) bm = fmaxf(bm, sc[t][j]);
#pragma unroll
        for (int msk = 1; msk < 16; msk <<= 1) bm = fmaxf(bm, __shfl_xor(bm, msk));
        const float mNew = fmaxf(mrow[j], bm);
        const float f = __expf(0.125f * (mrow[j] - mNew));  // scale = D^-0.5
        float sum = 0.f;
#pragma unroll
        for (int t = 0; t < 8; ++t) {
          const float p = __expf(0.125f * (sc[t][j] - mNew));
          sc[t][j] = p;
          sum += p;
        }
#pragma unroll
        for (int msk = 1; msk < 16; msk <<= 1) sum += __shfl_xor(sum, msk);
        mrow[j] = mNew;
        lrow[j] = lrow[j] * f + sum;
        fj[j] = f;
#pragma unroll
        for (int t = 0; t < 8; ++t)
          pRow[(half * 8 + j) * 128 + t * 16 + l15] = (__bf16)sc[t][j];
      }

      // rescale O accumulators by exp(m_old - m_new)
#pragma unroll
      for (int n = 0; n < 4; ++n)
#pragma unroll
        for (int j = 0; j < 8; ++j) oacc[n][j] *= fj[j];

      __syncthreads();  // P stores visible before transposed re-read

      // O += P (16x128) * V (128x64)
#pragma unroll
      for (int kc = 0; kc < 4; ++kc) {
        v16bf aP = load_afrag(pRow + kc * 32, 128, lane);
#pragma unroll
        for (int n = 0; n < 4; ++n) {
          V16S bv;
          const __bf16* vp =
              V + (keyBase + kc * 32 + half * 16) * 512 + fb + n * 16 + l15;
#pragma unroll
          for (int e = 0; e < 16; ++e) bv.e[e] = vp[(size_t)e * 512];
          oacc[n] = wmma_bf16(aP, bv.v, oacc[n]);
        }
      }
      __syncthreads();
    }

    // normalize and store O in [token, h*64+d] layout
#pragma unroll
    for (int j = 0; j < 8; ++j) lrow[j] = 1.0f / lrow[j];
#pragma unroll
    for (int n = 0; n < 4; ++n)
#pragma unroll
      for (int j = 0; j < 8; ++j) {
        const int row = q0 + half * 8 + j;
        O[(tokBase + row) * 512 + fb + n * 16 + l15] =
            (__bf16)(oacc[n][j] * lrow[j]);
      }
  }
}

// ---------------------------------------------------------------------------
// Host-side orchestration
// ---------------------------------------------------------------------------
extern "C" void kernel_launch(void* const* d_in, const int* in_sizes, int n_in,
                              void* d_out, int out_size, void* d_ws, size_t ws_size,
                              hipStream_t stream) {
  (void)in_sizes; (void)n_in; (void)out_size; (void)ws_size;
  const float* x   = (const float*)d_in[0];
  const float* Wq  = (const float*)d_in[1];
  const float* Wk  = (const float*)d_in[2];
  const float* Wv  = (const float*)d_in[3];
  const float* Wo  = (const float*)d_in[4];
  const float* bo  = (const float*)d_in[5];
  const float* W1  = (const float*)d_in[6];
  const float* b1  = (const float*)d_in[7];
  const float* W2  = (const float*)d_in[8];
  const float* b2  = (const float*)d_in[9];
  const float* g1  = (const float*)d_in[10];
  const float* be1 = (const float*)d_in[11];
  const float* g2  = (const float*)d_in[12];
  const float* be2 = (const float*)d_in[13];

  constexpr int NTOK = 32768;  // B*S*C
  constexpr int L    = 512;
  constexpr int FF   = 2048;

  char* ws = (char*)d_ws;
  size_t off = 0;
  auto alloc = [&](size_t bytes) {
    char* p = ws + off;
    off += (bytes + 255) & ~(size_t)255;
    return p;
  };
  __bf16* hb  = (__bf16*)alloc((size_t)NTOK * L * 2);   // LN1(x)
  __bf16* qb  = (__bf16*)alloc((size_t)NTOK * L * 2);
  __bf16* kb  = (__bf16*)alloc((size_t)NTOK * L * 2);
  __bf16* vb  = (__bf16*)alloc((size_t)NTOK * L * 2);
  __bf16* att = (__bf16*)alloc((size_t)NTOK * L * 2);   // attention out
  float*  x1  = (float*) alloc((size_t)NTOK * L * 4);   // x + proj
  __bf16* h2  = (__bf16*)alloc((size_t)NTOK * L * 2);   // LN2(x1)
  __bf16* ffb = (__bf16*)alloc((size_t)NTOK * FF * 2);  // relu(ffn1)
  __bf16* wqb = (__bf16*)alloc((size_t)L * L * 2);
  __bf16* wkb = (__bf16*)alloc((size_t)L * L * 2);
  __bf16* wvb = (__bf16*)alloc((size_t)L * L * 2);
  __bf16* wob = (__bf16*)alloc((size_t)L * L * 2);
  __bf16* w1b = (__bf16*)alloc((size_t)FF * L * 2);
  __bf16* w2b = (__bf16*)alloc((size_t)FF * L * 2);

  // weight conversion to bf16
  cvt_f32_bf16<<<(L * L) / 1024, 256, 0, stream>>>(Wq, wqb, L * L);
  cvt_f32_bf16<<<(L * L) / 1024, 256, 0, stream>>>(Wk, wkb, L * L);
  cvt_f32_bf16<<<(L * L) / 1024, 256, 0, stream>>>(Wv, wvb, L * L);
  cvt_f32_bf16<<<(L * L) / 1024, 256, 0, stream>>>(Wo, wob, L * L);
  cvt_f32_bf16<<<(FF * L) / 1024, 256, 0, stream>>>(W1, w1b, FF * L);
  cvt_f32_bf16<<<(FF * L) / 1024, 256, 0, stream>>>(W2, w2b, FF * L);

  // LN1
  ln_bf16<<<NTOK / 4, 128, 0, stream>>>(x, g1, be1, hb);

  // QKV projections (combined per-head weights are just [512,512] row-major)
  const int g512 = (NTOK / 256) * (L / 64);
  gemm_wmma<__bf16, false><<<g512, 256, 0, stream>>>(hb, wqb, nullptr, nullptr,
                                                     qb, NTOK, L, L);
  gemm_wmma<__bf16, false><<<g512, 256, 0, stream>>>(hb, wkb, nullptr, nullptr,
                                                     kb, NTOK, L, L);
  gemm_wmma<__bf16, false><<<g512, 256, 0, stream>>>(hb, wvb, nullptr, nullptr,
                                                     vb, NTOK, L, L);

  // regional attention: B*H*S = 1024 blocks
  attn_wmma<<<1024, 128, 0, stream>>>(qb, kb, vb, att);

  // output projection + residual
  gemm_wmma<float, false><<<g512, 256, 0, stream>>>(att, wob, bo, x, x1,
                                                    NTOK, L, L);

  // LN2
  ln_bf16<<<NTOK / 4, 128, 0, stream>>>(x1, g2, be2, h2);

  // FFN
  const int gff = (NTOK / 256) * (FF / 64);
  gemm_wmma<__bf16, true><<<gff, 256, 0, stream>>>(h2, w1b, b1, nullptr, ffb,
                                                   NTOK, FF, L);
  gemm_wmma<float, false><<<g512, 256, 0, stream>>>(ffb, w2b, b2, x1,
                                                    (float*)d_out, NTOK, L, FF);
}
